// SelfAttention_62723702391426
// MI455X (gfx1250) — matrix operand order repaired and checked by reading, not compiled, not run on hardware
//
#include <hip/hip_runtime.h>
#include <hip/hip_bf16.h>

typedef unsigned short u16;
typedef unsigned int   u32;
typedef unsigned long long u64;
typedef __attribute__((ext_vector_type(16))) __bf16 v16bf;
typedef __attribute__((ext_vector_type(8)))  float  v8f;
typedef __attribute__((ext_vector_type(4)))  unsigned int v4u;
typedef __attribute__((ext_vector_type(8)))  int v8i_;
typedef __attribute__((ext_vector_type(4)))  int v4i_;

union Frag16 { v16bf v; u32 w[8]; };

__device__ __forceinline__ u16 f32_to_bf16(float f) {
    u32 u = __float_as_uint(f);
    u += 0x7FFFu + ((u >> 16) & 1u);   // round to nearest even
    return (u16)(u >> 16);
}

// ---------------------------------------------------------------------------
// Tensor Data Mover: 2-D tile (bf16 elements) global -> LDS, with optional
// LDS row padding. D# layout per CDNA5 ISA 8.3/8.4. One call per wave.
// 6-arg builtin form (amdgpu-toolchain / clang-23).
//   tileD0   : contiguous elements per row      tileD1 : rows
//   strideEl : row stride in elements           padI/padA : D# pad codes
// ---------------------------------------------------------------------------
__device__ __forceinline__ void tdm_load_2d_bf16(u32 ldsAddr, const void* gptr,
                                                 u32 tileD0, u32 tileD1,
                                                 u64 strideEl,
                                                 u32 padI, u32 padA, bool padEn) {
    u64 ga = (u64)gptr;
    v4u g0;
    g0.x = 1u;                                            // count=1, user mode
    g0.y = ldsAddr;                                       // lds_addr [63:32]
    g0.z = (u32)ga;                                       // global_addr lo
    g0.w = (u32)((ga >> 32) & 0x1FFFFFFu) | (2u << 30);   // addr hi + type=2
    v8i_ g1;
    u32 d0 = (1u << 16);                                  // data_size = 2 bytes
    if (padEn) d0 |= (1u << 20) | (padI << 22) | (padA << 25);
    g1[0] = (int)d0;                                      // wg_mask=0
    g1[1] = (int)((tileD0 & 0xFFFFu) << 16);              // tensor_dim0 = tileD0
    g1[2] = (int)((tileD0 >> 16) | ((tileD1 & 0xFFFFu) << 16)); // tensor_dim1
    g1[3] = (int)((tileD1 >> 16) | (tileD0 << 16));       // tile_dim0
    g1[4] = (int)(tileD1 & 0xFFFFu);                      // tile_dim1, tile_dim2=0
    g1[5] = (int)(u32)(strideEl & 0xFFFFFFFFu);           // dim0_stride lo
    g1[6] = (int)(u32)((strideEl >> 32) & 0xFFFFu);       // dim0_stride hi, dim1_stride=0
    g1[7] = 0;
    v4i_ z4 = {0, 0, 0, 0};
    v8i_ z8 = {0, 0, 0, 0, 0, 0, 0, 0};
    __builtin_amdgcn_tensor_load_to_lds(g0, g1, z4, z4, z8, 0);
}

// ---------------------------------------------------------------------------
// Elementwise f32 -> bf16 cast (4 elems/thread)
// ---------------------------------------------------------------------------
__global__ __launch_bounds__(256) void cast_f32_to_bf16(const float* __restrict__ in,
                                                        u16* __restrict__ out, long n) {
    long i = ((long)blockIdx.x * blockDim.x + threadIdx.x) * 4;
    if (i >= n) return;
    float4 f = *(const float4*)(in + i);
    u16 r[4];
    r[0] = f32_to_bf16(f.x); r[1] = f32_to_bf16(f.y);
    r[2] = f32_to_bf16(f.z); r[3] = f32_to_bf16(f.w);
    *(uint2*)(out + i) = *(const uint2*)r;
}

// ---------------------------------------------------------------------------
// bf16 GEMM: C[M,N] = A[M,K] * B[K,N], f32 accumulate.
// Block: 256 thr = 8 waves; block tile 128x128; wave tile 32x64; K step 32.
// A tile (row-major) filled by TDM; B tile transposed by thread scatter.
// ---------------------------------------------------------------------------
template<bool F32OUT>
__global__ __launch_bounds__(256) void gemm_bf16(const u16* __restrict__ A,
                                                 const u16* __restrict__ Bm,
                                                 void* __restrict__ Cv,
                                                 int M, int N, int K) {
    __shared__ __align__(16) u16 As[128 * 40];   // [m][k] stride 40 (TDM pad)
    __shared__ __align__(16) u16 Bt[128 * 40];   // [n][k] stride 40

    const int t = threadIdx.x;
    const int lane = t & 31, w = t >> 5;
    const int g = lane >> 4, nl = lane & 15;
    const int waveRow = w >> 1, waveCol = w & 1;
    const int m0 = blockIdx.y * 128, n0 = blockIdx.x * 128;
    const u32 asAddr = (u32)(size_t)&As[0];
    (void)M;

    v8f acc[2][4];
    const v8f zero = {0.f,0.f,0.f,0.f,0.f,0.f,0.f,0.f};
    #pragma unroll
    for (int mt = 0; mt < 2; mt++)
        #pragma unroll
        for (int nt = 0; nt < 4; nt++) acc[mt][nt] = zero;

    for (int k0 = 0; k0 < K; k0 += 32) {
        __syncthreads();
        if (t < 32) {   // wave 0: TDM DMA of the 128x32 A tile (row = 64B, pad 16B)
            tdm_load_2d_bf16(asAddr, A + (size_t)m0 * K + k0,
                             /*tileD0=*/32, /*tileD1=*/128, (u64)K,
                             /*padI=*/3, /*padA=*/3, true);
        }
        {   // all threads: B tile 32x128 -> Bt[n][k]
            int kk = t & 31, ng = t >> 5;
            const u16* src = Bm + (size_t)(k0 + kk) * N + n0 + ng * 16;
            u16 tmp[16];
            *(uint4*)tmp       = *(const uint4*)src;
            *(uint4*)(tmp + 8) = *(const uint4*)(src + 8);
            #pragma unroll
            for (int i = 0; i < 16; i++) Bt[(ng * 16 + i) * 40 + kk] = tmp[i];
        }
        if (t < 32) __builtin_amdgcn_s_wait_tensorcnt(0);
        __syncthreads();

        Frag16 af[2], bfr[4];
        #pragma unroll
        for (int mt = 0; mt < 2; mt++)
            #pragma unroll
            for (int i = 0; i < 8; i++) {
                int kk = ((i >> 2) << 4) + ((i & 3) << 1) + (g << 3);  // A layout
                af[mt].w[i] = *(const u32*)&As[(waveRow * 32 + mt * 16 + nl) * 40 + kk];
            }
        #pragma unroll
        for (int nt = 0; nt < 4; nt++)
            #pragma unroll
            for (int i = 0; i < 8; i++) {
                int kk = ((i >> 2) << 4) + ((i & 3) << 1) + (g << 3);  // B layout
                bfr[nt].w[i] = *(const u32*)&Bt[(waveCol * 64 + nt * 16 + nl) * 40 + kk];
            }
        #pragma unroll
        for (int mt = 0; mt < 2; mt++)
            #pragma unroll
            for (int nt = 0; nt < 4; nt++)
                acc[mt][nt] = __builtin_amdgcn_wmma_f32_16x16x32_bf16(
                    false, af[mt].v, false, bfr[nt].v, (short)0, acc[mt][nt], false, false);
    }

    const int row0 = m0 + waveRow * 32, col0 = n0 + waveCol * 64;
    #pragma unroll
    for (int mt = 0; mt < 2; mt++)
        #pragma unroll
        for (int nt = 0; nt < 4; nt++) {
            int col = col0 + nt * 16 + nl;
            #pragma unroll
            for (int r = 0; r < 8; r++) {
                int row = row0 + mt * 16 + (g << 3) + r;   // C/D layout: M = r + 8g
                if (F32OUT) ((float*)Cv)[(size_t)row * N + col] = acc[mt][nt][r];
                else        ((u16*)Cv)[(size_t)row * N + col]  = f32_to_bf16(acc[mt][nt][r]);
            }
        }
}

// ---------------------------------------------------------------------------
// Flash attention (causal). qkv: bf16 [B,T,3C] ([q|k|v], head h at +h*128).
// Block: 256 thr = 8 waves, 128 q rows (16 per wave). Key tiles of 32.
// Q and K tiles DMA'd by TDM; V transposed by thread scatter.
// ---------------------------------------------------------------------------
__global__ __launch_bounds__(256) void attn_kernel(const u16* __restrict__ qkv,
                                                   u16* __restrict__ y) {
    constexpr int T = 2048, C3 = 6144, C = 2048, HS = 128;
    __shared__ __align__(16) u16 Qs[128 * 136];
    __shared__ __align__(16) u16 Ks[32 * 136];
    __shared__ __align__(16) u16 Vt[128 * 40];       // [d][key]
    __shared__ __align__(16) u16 Pld[8 * 16 * 40];   // per-wave P restage

    const int t = threadIdx.x, lane = t & 31, w = t >> 5;
    const int g = lane >> 4, nl = lane & 15;
    const int qBase = blockIdx.x * 128, h = blockIdx.y, b = blockIdx.z;
    const size_t baseQ = (size_t)b * T * C3 + (size_t)h * HS;
    const u32 qsAddr = (u32)(size_t)&Qs[0];
    const u32 ksAddr = (u32)(size_t)&Ks[0];

    if (t < 32) {   // TDM: 128x128 Q tile (row 256B, pad 16B -> stride 136)
        tdm_load_2d_bf16(qsAddr, qkv + baseQ + (size_t)qBase * C3,
                         /*tileD0=*/128, /*tileD1=*/128, (u64)C3,
                         /*padI=*/5, /*padA=*/3, true);
        __builtin_amdgcn_s_wait_tensorcnt(0);
    }
    __syncthreads();

    Frag16 qf[4];
    #pragma unroll
    for (int kb = 0; kb < 4; kb++)
        #pragma unroll
        for (int i = 0; i < 8; i++) {
            int kk = kb * 32 + ((i >> 2) << 4) + ((i & 3) << 1) + (g << 3);
            qf[kb].w[i] = *(const u32*)&Qs[(w * 16 + nl) * 136 + kk];
        }

    const v8f zero = {0.f,0.f,0.f,0.f,0.f,0.f,0.f,0.f};
    v8f acc[8];
    float mrow[8], lrow[8];
    #pragma unroll
    for (int j = 0; j < 8; j++) acc[j] = zero;
    #pragma unroll
    for (int r = 0; r < 8; r++) { mrow[r] = -3.0e38f; lrow[r] = 0.f; }
    const float scale = 0.08838834764831845f;   // 1/sqrt(128)
    const int nkt = qBase / 32 + 4;             // causal: keys 0 .. qBase+127

    for (int kt = 0; kt < nkt; kt++) {
        const int kBase = kt * 32;
        __syncthreads();
        if (t < 32) {   // TDM: 32x128 K tile
            tdm_load_2d_bf16(ksAddr, qkv + baseQ + 2048 + (size_t)kBase * C3,
                             /*tileD0=*/128, /*tileD1=*/32, (u64)C3,
                             /*padI=*/5, /*padA=*/3, true);
        }
        {   // all threads: V tile 32x128 stored transposed [d][key]
            int kr = t & 31, ds = (t >> 5) * 16;
            const u16* src = qkv + baseQ + 4096 + (size_t)(kBase + kr) * C3 + ds;
            u16 tmp[16];
            *(uint4*)tmp       = *(const uint4*)src;
            *(uint4*)(tmp + 8) = *(const uint4*)(src + 8);
            #pragma unroll
            for (int i = 0; i < 16; i++) Vt[(ds + i) * 40 + kr] = tmp[i];
        }
        if (t < 32) __builtin_amdgcn_s_wait_tensorcnt(0);
        __syncthreads();

        // S = Q * K^T : two 16x16 f32 accums (key cols 0..15, 16..31)
        v8f s[2]; s[0] = zero; s[1] = zero;
        #pragma unroll
        for (int ct = 0; ct < 2; ct++)
            #pragma unroll
            for (int kb = 0; kb < 4; kb++) {
                Frag16 bfr;
                #pragma unroll
                for (int i = 0; i < 8; i++) {
                    int dd = kb * 32 + ((i >> 2) << 4) + ((i & 3) << 1) + (g << 3);
                    bfr.w[i] = *(const u32*)&Ks[(ct * 16 + nl) * 136 + dd];
                }
                s[ct] = __builtin_amdgcn_wmma_f32_16x16x32_bf16(
                    false, qf[kb].v, false, bfr.v, (short)0, s[ct], false, false);
            }

        // scale + causal mask + online softmax
        const int qIdx = qBase + w * 16 + (g << 3);
        float t0[8], t1[8], mnew[8], alpha[8];
        #pragma unroll
        for (int r = 0; r < 8; r++) {
            int q = qIdx + r;
            float v0 = s[0][r] * scale; if (kBase + nl      > q) v0 = -3.0e38f;
            float v1 = s[1][r] * scale; if (kBase + 16 + nl > q) v1 = -3.0e38f;
            t0[r] = v0; t1[r] = v1;
        }
        #pragma unroll
        for (int r = 0; r < 8; r++) {
            float rm = fmaxf(t0[r], t1[r]);
            #pragma unroll
            for (int off = 1; off < 16; off <<= 1) rm = fmaxf(rm, __shfl_xor(rm, off, 32));
            mnew[r]  = fmaxf(mrow[r], rm);
            alpha[r] = __expf(mrow[r] - mnew[r]);
        }
        #pragma unroll
        for (int r = 0; r < 8; r++) {
            float p0 = (t0[r] <= -1.0e37f) ? 0.f : __expf(t0[r] - mnew[r]);
            float p1 = (t1[r] <= -1.0e37f) ? 0.f : __expf(t1[r] - mnew[r]);
            float rs = p0 + p1;
            #pragma unroll
            for (int off = 1; off < 16; off <<= 1) rs += __shfl_xor(rs, off, 32);
            lrow[r] = lrow[r] * alpha[r] + rs;
            mrow[r] = mnew[r];
            u16* pw = &Pld[(w * 16 + (g << 3) + r) * 40];
            pw[nl]      = f32_to_bf16(p0);
            pw[16 + nl] = f32_to_bf16(p1);
        }
        #pragma unroll
        for (int j = 0; j < 8; j++)
            #pragma unroll
            for (int r = 0; r < 8; r++) acc[j][r] *= alpha[r];

        // restaged P -> A-fragment (wave-private LDS, DScnt-ordered)
        Frag16 pf;
        #pragma unroll
        for (int i = 0; i < 8; i++) {
            int kk = ((i >> 2) << 4) + ((i & 3) << 1) + (g << 3);
            pf.w[i] = *(const u32*)&Pld[(w * 16 + nl) * 40 + kk];
        }
        // O += P * V
        #pragma unroll
        for (int j = 0; j < 8; j++) {
            Frag16 vf;
            #pragma unroll
            for (int i = 0; i < 8; i++) {
                int kk = ((i >> 2) << 4) + ((i & 3) << 1) + (g << 3);
                vf.w[i] = *(const u32*)&Vt[(j * 16 + nl) * 40 + kk];
            }
            acc[j] = __builtin_amdgcn_wmma_f32_16x16x32_bf16(
                false, pf.v, false, vf.v, (short)0, acc[j], false, false);
        }
    }

    // normalize + store y (bf16, [B,T,C] with head offset)
    const size_t yBase = (size_t)b * T * C + (size_t)h * HS;
    #pragma unroll
    for (int r = 0; r < 8; r++) {
        float inv = 1.0f / lrow[r];
        int row = qBase + w * 16 + (g << 3) + r;
        #pragma unroll
        for (int j = 0; j < 8; j++)
            y[yBase + (size_t)row * C + j * 16 + nl] = f32_to_bf16(acc[j][r] * inv);
    }
}

// ---------------------------------------------------------------------------
extern "C" void kernel_launch(void* const* d_in, const int* in_sizes, int n_in,
                              void* d_out, int out_size, void* d_ws, size_t ws_size,
                              hipStream_t stream) {
    (void)in_sizes; (void)n_in; (void)out_size; (void)ws_size;
    const float* x      = (const float*)d_in[0];
    const float* w_qkv  = (const float*)d_in[1];
    const float* w_proj = (const float*)d_in[2];
    float* out = (float*)d_out;

    const long B = 4, T = 2048, C = 2048, C3 = 6144;
    const long nX = B * T * C, nWq = C * C3, nWp = C * C, nQKV = B * T * C3;

    char* ws = (char*)d_ws;
    size_t o = 0;
    u16* xb    = (u16*)(ws + o); o += (size_t)nX   * 2;
    u16* wqkvb = (u16*)(ws + o); o += (size_t)nWq  * 2;
    u16* wprojb= (u16*)(ws + o); o += (size_t)nWp  * 2;
    u16* qkvb  = (u16*)(ws + o); o += (size_t)nQKV * 2;
    u16* yb    = (u16*)(ws + o);

    auto blocks4 = [](long n) { return (unsigned)((n / 4 + 255) / 256); };
    cast_f32_to_bf16<<<blocks4(nX),  256, 0, stream>>>(x,      xb,     nX);
    cast_f32_to_bf16<<<blocks4(nWq), 256, 0, stream>>>(w_qkv,  wqkvb,  nWq);
    cast_f32_to_bf16<<<blocks4(nWp), 256, 0, stream>>>(w_proj, wprojb, nWp);

    // qkv = x @ w_qkv : M=8192, N=6144, K=2048, bf16 out
    gemm_bf16<false><<<dim3(6144 / 128, 8192 / 128), 256, 0, stream>>>(
        xb, wqkvb, (void*)qkvb, 8192, 6144, 2048);

    // flash attention: grid (qtiles=16, heads=16, batch=4)
    attn_kernel<<<dim3(16, 16, 4), 256, 0, stream>>>(qkvb, yb);

    // out = y @ w_proj : M=8192, N=2048, K=2048, f32 out
    gemm_bf16<true><<<dim3(2048 / 128, 8192 / 128), 256, 0, stream>>>(
        yb, wprojb, (void*)out, 8192, 2048, 2048);
}
